// SCformer_58162447123011
// MI455X (gfx1250) — matrix operand order repaired and checked by reading, not compiled, not run on hardware
//
#include <hip/hip_runtime.h>
#include <math.h>
#include <stdint.h>

typedef __attribute__((ext_vector_type(16))) _Float16 v16h;
typedef __attribute__((ext_vector_type(8)))  float    v8f;
typedef __attribute__((__vector_size__(4 * sizeof(int)))) int v4i_;

// Problem constants
#define B_    4
#define N_    4096
#define C_    32
#define K_    16
#define H_    260
#define W_    346
#define HW_   (H_*W_)
#define CIN_  34          // p, 1-p, 32 features
#define CINP_ 64          // padded channels for WMMA K
#define COUT_ 96          // 3*C
#define NTAP_ 81
#define RAD_  (5.0f/260.0f)
#define EPS_  1e-5f
#define MT_   4           // M tiles (of 16 events) per block
#define EVB_  (MT_*16)    // 64 events per block

// Workspace layout (bytes)
static constexpr size_t SZ_IDX   = (size_t)B_*N_*K_*4;
static constexpr size_t SZ_STATS = 256;
static constexpr size_t SZ_IMG32 = (size_t)B_*HW_*CIN_*4;
static constexpr size_t SZ_IMG16 = (size_t)B_*HW_*CINP_*2;
static constexpr size_t SZ_WT    = (size_t)NTAP_*COUT_*CINP_*2;
static constexpr size_t OFF_IDX   = 0;
static constexpr size_t OFF_STATS = OFF_IDX   + SZ_IDX;
static constexpr size_t OFF_IMG32 = OFF_STATS + SZ_STATS;
static constexpr size_t OFF_IMG16 = OFF_IMG32 + SZ_IMG32;
static constexpr size_t OFF_WT    = OFF_IMG16 + SZ_IMG16;
static constexpr size_t OFF_ST    = OFF_WT    + SZ_WT;

// -------- async global -> LDS (CDNA5 GLOBAL_LOAD_ASYNC_TO_LDS_B128, ASYNCcnt)
#if __has_builtin(__builtin_amdgcn_global_load_async_to_lds_b128)
__device__ __forceinline__ void async_cp16(void* dst_lds, const void* src_g) {
    __builtin_amdgcn_global_load_async_to_lds_b128(
        (__attribute__((address_space(1))) v4i_*)(uintptr_t)src_g,
        (__attribute__((address_space(3))) v4i_*)(uint32_t)(uintptr_t)dst_lds,
        0, 0);
}
#else
__device__ __forceinline__ void async_cp16(void* dst_lds, const void* src_g) {
    asm volatile("global_load_async_to_lds_b128 %0, %1, off"
                 :: "v"((uint32_t)(uintptr_t)dst_lds),
                    "v"((uint64_t)(uintptr_t)src_g)
                 : "memory");
}
#endif
#if __has_builtin(__builtin_amdgcn_s_wait_asynccnt)
#define AWAIT(n) __builtin_amdgcn_s_wait_asynccnt(n)
#else
#define AWAIT(n) asm volatile("s_wait_asynccnt %0" :: "n"(n) : "memory")
#endif

// ---------------------------------------------------------------- utilities
__global__ void k_zero(float* __restrict__ p, int n) {
    for (int i = blockIdx.x*blockDim.x + threadIdx.x; i < n; i += gridDim.x*blockDim.x)
        p[i] = 0.0f;
}

// ---------------------------------------------------------------- ball query
__global__ __launch_bounds__(256) void k_ballquery(const float* __restrict__ xytp,
                                                   int* __restrict__ idx) {
    __shared__ float scx[256], scy[256];
    const int b     = blockIdx.x / (N_/256);
    const int n     = (blockIdx.x % (N_/256))*256 + threadIdx.x;
    const float qx  = xytp[((size_t)b*N_ + n)*4 + 1];
    const float qy  = xytp[((size_t)b*N_ + n)*4 + 2];
    int* my = idx + ((size_t)b*N_ + n)*K_;
    int cnt = 0;
    for (int tb = 0; tb < N_; tb += 256) {
        __syncthreads();
        const int j = tb + threadIdx.x;
        scx[threadIdx.x] = xytp[((size_t)b*N_ + j)*4 + 1];
        scy[threadIdx.x] = xytp[((size_t)b*N_ + j)*4 + 2];
        __syncthreads();
        if (cnt < K_) {
            for (int jj = 0; jj < 256 && cnt < K_; ++jj) {
                const float dx = qx - scx[jj], dy = qy - scy[jj];
                if (dx*dx + dy*dy < RAD_*RAD_) { my[cnt++] = tb + jj; }
            }
        }
    }
    for (; cnt < K_; ++cnt) my[cnt] = n;
}

// ---------------------------------------------------------------- BN stats
__global__ __launch_bounds__(256) void k_stats(const float* __restrict__ xytp,
                                               const int* __restrict__ idx,
                                               const float* __restrict__ w1,
                                               const float* __restrict__ b1,
                                               float* __restrict__ stats) {
    const int e = blockIdx.x*blockDim.x + threadIdx.x;
    const int b = e / N_;
    const float qx = xytp[(size_t)e*4 + 1];
    const float qy = xytp[(size_t)e*4 + 2];
    const float w00 = w1[0], w01 = w1[1], w10 = w1[2], w11 = w1[3];
    const float b10 = b1[0], b11 = b1[1];
    float s0 = 0.f, s1 = 0.f, ss0 = 0.f, ss1 = 0.f;
    #pragma unroll
    for (int k = 0; k < K_; ++k) {
        const int j = idx[(size_t)e*K_ + k];
        const float dx = qx - xytp[((size_t)b*N_ + j)*4 + 1];
        const float dy = qy - xytp[((size_t)b*N_ + j)*4 + 2];
        const float h0 = dx*w00 + dy*w10 + b10;
        const float h1 = dx*w01 + dy*w11 + b11;
        s0 += h0; s1 += h1; ss0 += h0*h0; ss1 += h1*h1;
    }
    for (int off = 16; off; off >>= 1) {
        s0  += __shfl_xor(s0,  off, 32);
        s1  += __shfl_xor(s1,  off, 32);
        ss0 += __shfl_xor(ss0, off, 32);
        ss1 += __shfl_xor(ss1, off, 32);
    }
    if ((threadIdx.x & 31) == 0) {
        atomicAdd(&stats[0], s0);  atomicAdd(&stats[1], s1);
        atomicAdd(&stats[2], ss0); atomicAdd(&stats[3], ss1);
    }
}

// ---------------------------------------------------------------- scatter
__global__ void k_scatter(const float* __restrict__ xytp,
                          const float* __restrict__ feats,
                          float* __restrict__ img32) {
    const int g = blockIdx.x*blockDim.x + threadIdx.x;
    if (g >= B_*N_*CIN_) return;
    const int c = g % CIN_;
    const int e = g / CIN_;
    const int b = e / N_;
    const float* ev = xytp + (size_t)e*4;
    int yy = (int)rintf(ev[2]*H_); yy = yy < 0 ? 0 : (yy > H_-1 ? H_-1 : yy);
    int xx = (int)rintf(ev[1]*W_); xx = xx < 0 ? 0 : (xx > W_-1 ? W_-1 : xx);
    float v;
    if      (c == 0) v = ev[3];
    else if (c == 1) v = 1.0f - ev[3];
    else             v = feats[(size_t)e*C_ + (c - 2)];
    atomicAdd(&img32[(((size_t)b*H_ + yy)*W_ + xx)*CIN_ + c], v);
}

// ---------------------------------------------------------------- f32 -> f16 (pad 34->64)
__global__ void k_convert(const float* __restrict__ img32, _Float16* __restrict__ img16) {
    const int g = blockIdx.x*blockDim.x + threadIdx.x;
    if (g >= B_*HW_*CINP_) return;
    const int c   = g & (CINP_-1);
    const int pix = g >> 6;
    _Float16 v = (_Float16)0.0f;
    if (c < CIN_) v = (_Float16)img32[(size_t)pix*CIN_ + c];
    img16[g] = v;
}

// ---------------------------------------------------------------- weight transpose
__global__ void k_wprep(const float* __restrict__ conv_w, _Float16* __restrict__ wt) {
    const int g = blockIdx.x*blockDim.x + threadIdx.x;
    if (g >= NTAP_*COUT_*CINP_) return;
    const int i = g & (CINP_-1);
    const int r = g >> 6;
    const int o = r % COUT_;
    const int t = r / COUT_;
    _Float16 v = (_Float16)0.0f;
    if (i < CIN_) v = (_Float16)conv_w[((size_t)t*CIN_ + i)*COUT_ + o];
    wt[((size_t)t*COUT_ + o)*CINP_ + i] = v;
}

// ---------------------------------------------------------------- sparse 9x9 conv via WMMA
// 64 events/block (4 M-tiles), 6 waves each owning one 16-wide cout tile.
// Double-buffered async global->LDS staging overlapped with WMMA compute.
__global__ __launch_bounds__(192) void k_conv(const float* __restrict__ xytp,
                                              const _Float16* __restrict__ img16,
                                              const _Float16* __restrict__ wt,
                                              const float* __restrict__ conv_b,
                                              float* __restrict__ st) {
    __shared__ alignas(16) _Float16 sA[2][EVB_*CINP_];    // 2 x 8 KB
    __shared__ alignas(16) _Float16 sB[2][COUT_*CINP_];   // 2 x 12 KB
    __shared__ int sEy[EVB_], sEx[EVB_], sEb[EVB_];

    const int tid = threadIdx.x;
    if (tid < EVB_) {
        const int e = blockIdx.x*EVB_ + tid;
        const float* ev = xytp + (size_t)e*4;
        int yy = (int)rintf(ev[2]*H_); yy = yy < 0 ? 0 : (yy > H_-1 ? H_-1 : yy);
        int xx = (int)rintf(ev[1]*W_); xx = xx < 0 ? 0 : (xx > W_-1 ? W_-1 : xx);
        sEy[tid] = yy; sEx[tid] = xx; sEb[tid] = e / N_;
    }
    const int lane = tid & 31;
    const int nt   = tid >> 5;           // wave id == cout tile, 0..5
    const int row  = lane & 15;
    const int hi   = lane >> 4;

    v8f acc[MT_];
    const float bias = conv_b[nt*16 + row];
    #pragma unroll
    for (int mt = 0; mt < MT_; ++mt)
        #pragma unroll
        for (int i = 0; i < 8; ++i) acc[mt][i] = bias;

    __syncthreads();

    // Per-thread staging state (registers, fixed across taps).
    // B: 768 b128 = 4 per thread.  A: 512 b128 = 4 per thread for tid<128.
    int ey[4], ex[4], eb[4];
    if (tid < 128) {
        #pragma unroll
        for (int r = 0; r < 4; ++r) {
            const int evi = (tid >> 3) + 16*r;
            ey[r] = sEy[evi]; ex[r] = sEx[evi]; eb[r] = sEb[evi];
        }
    }
    const int ccOff = (tid & 7)*8;                 // A channel sub-block
    const _Float16* zsrc = img16 + 48;             // 16 guaranteed-zero halfs

    // tap issue: stage A+B for `tap` into buffer `buf`
    auto issue = [&](int tap, int buf) {
        const _Float16* wtap = wt + (size_t)tap*COUT_*CINP_;
        _Float16* sBb = sB[buf];
        #pragma unroll
        for (int r = 0; r < 4; ++r) {
            const int j = (tid + 192*r)*8;
            async_cp16(sBb + j, wtap + j);
        }
        if (tid < 128) {
            const int dy = tap/9 - 4, dx = tap%9 - 4;
            _Float16* sAb = sA[buf];
            #pragma unroll
            for (int r = 0; r < 4; ++r) {
                const int yy = ey[r] + dy, xx = ex[r] + dx;
                const _Float16* src = zsrc;
                if (yy >= 0 && yy < H_ && xx >= 0 && xx < W_)
                    src = img16 + (((size_t)eb[r]*H_ + yy)*W_ + xx)*CINP_ + ccOff;
                async_cp16(sAb + (tid + 128*r)*8, src);
            }
        }
    };

    issue(0, 0);
    for (int tap = 0; tap < NTAP_; ++tap) {
        const int buf = tap & 1;
        if (tap + 1 < NTAP_) {
            issue(tap + 1, buf ^ 1);
            if (tid < 128) { AWAIT(8); } else { AWAIT(4); }   // uniform per wave
        } else {
            AWAIT(0);
        }
        __syncthreads();

        const _Float16* sAb = sA[buf];
        const _Float16* sBb = sB[buf];
        #pragma unroll
        for (int kc = 0; kc < 2; ++kc) {
            union { v16h v; uint4 q[2]; } bf;
            const _Float16* bp = sBb + (nt*16 + row)*CINP_ + kc*32 + hi*8;
            bf.q[0] = *(const uint4*)(bp);
            bf.q[1] = *(const uint4*)(bp + 16);
            #pragma unroll
            for (int mt = 0; mt < MT_; ++mt) {
                union { v16h v; uint4 q[2]; } af;
                const _Float16* ap = sAb + (mt*16 + row)*CINP_ + kc*32 + hi*8;
                af.q[0] = *(const uint4*)(ap);
                af.q[1] = *(const uint4*)(ap + 16);
                acc[mt] = __builtin_amdgcn_wmma_f32_16x16x32_f16(
                              false, af.v, false, bf.v, (short)0, acc[mt], false, false);
            }
        }
        __syncthreads();
    }

    // C/D layout: lanes 0-15: M=vgpr, N=lane; lanes 16-31: M=8+vgpr, N=lane-16
    #pragma unroll
    for (int mt = 0; mt < MT_; ++mt)
        #pragma unroll
        for (int v = 0; v < 8; ++v) {
            const int M = mt*16 + hi*8 + v;
            st[((size_t)blockIdx.x*EVB_ + M)*COUT_ + nt*16 + row] = acc[mt][v];
        }
}

// ---------------------------------------------------------------- attention epilogue
__global__ __launch_bounds__(256) void k_attn(const float* __restrict__ xytp,
                                              const int* __restrict__ idx,
                                              const float* __restrict__ stats,
                                              const float* __restrict__ w1,
                                              const float* __restrict__ b1,
                                              const float* __restrict__ bng,
                                              const float* __restrict__ bnb,
                                              const float* __restrict__ w2,
                                              const float* __restrict__ b2,
                                              const float* __restrict__ st,
                                              const float* __restrict__ lng,
                                              const float* __restrict__ lnb,
                                              float* __restrict__ out) {
    const int wid  = (blockIdx.x*blockDim.x + threadIdx.x) >> 5;
    const int lane = threadIdx.x & 31;
    if (wid >= B_*N_) return;
    const int e = wid;
    const int b = e / N_;

    const float inv_cnt = 1.0f / (float)((size_t)B_*N_*K_);
    const float mu0 = stats[0]*inv_cnt, mu1 = stats[1]*inv_cnt;
    const float rs0 = rsqrtf(stats[2]*inv_cnt - mu0*mu0 + EPS_) * bng[0];
    const float rs1 = rsqrtf(stats[3]*inv_cnt - mu1*mu1 + EPS_) * bng[1];
    const float bb0 = bnb[0], bb1 = bnb[1];
    const float w00 = w1[0], w01 = w1[1], w10 = w1[2], w11 = w1[3];
    const float b10 = b1[0], b11 = b1[1];
    const float w2a = w2[lane], w2b = w2[C_ + lane], b2c = b2[lane];
    const float lg = lng[lane], lb = lnb[lane];

    const float* ev = xytp + (size_t)e*4;
    const float qx = ev[1], qy = ev[2];
    const float varphi = st[(size_t)e*COUT_ + lane];

    float qv[K_], av[K_];
    #pragma unroll
    for (int k = 0; k < K_; ++k) {
        const int j = idx[(size_t)e*K_ + k];
        const float* nb = xytp + ((size_t)b*N_ + j)*4;
        const float dx = qx - nb[1], dy = qy - nb[2];
        float h0 = (dx*w00 + dy*w10 + b10 - mu0)*rs0 + bb0;
        float h1 = (dx*w01 + dy*w11 + b11 - mu1)*rs1 + bb1;
        h0 = fmaxf(h0, 0.f); h1 = fmaxf(h1, 0.f);
        const float delta = h0*w2a + h1*w2b + b2c;
        const float* srow = st + ((size_t)b*N_ + j)*COUT_;
        float q = varphi - srow[C_ + lane] + delta;
        av[k]   = srow[2*C_ + lane] + delta;
        float m = q;
        for (int off = 16; off; off >>= 1) m += __shfl_xor(m, off, 32);
        m *= (1.0f/32.0f);
        const float d = q - m;
        float vv = d*d;
        for (int off = 16; off; off >>= 1) vv += __shfl_xor(vv, off, 32);
        vv *= (1.0f/32.0f);
        qv[k] = d * rsqrtf(vv + EPS_) * lg + lb;
    }
    const float scale = 0.1767766952966369f;   // 1/sqrt(32)
    float mx = -INFINITY;
    #pragma unroll
    for (int k = 0; k < K_; ++k) mx = fmaxf(mx, qv[k]*scale);
    float Z = 0.f;
    #pragma unroll
    for (int k = 0; k < K_; ++k) { qv[k] = __expf(qv[k]*scale - mx); Z += qv[k]; }
    const float rZ = 1.0f / Z;
    float o = 0.f;
    #pragma unroll
    for (int k = 0; k < K_; ++k) o += qv[k]*rZ*av[k];
    out[(size_t)e*C_ + lane] = o;
}

// ---------------------------------------------------------------- launch
extern "C" void kernel_launch(void* const* d_in, const int* in_sizes, int n_in,
                              void* d_out, int out_size, void* d_ws, size_t ws_size,
                              hipStream_t stream) {
    const float* xytp   = (const float*)d_in[0];
    const float* feats  = (const float*)d_in[1];
    const float* w1     = (const float*)d_in[2];
    const float* b1     = (const float*)d_in[3];
    const float* bng    = (const float*)d_in[4];
    const float* bnb    = (const float*)d_in[5];
    const float* w2     = (const float*)d_in[6];
    const float* b2     = (const float*)d_in[7];
    const float* conv_w = (const float*)d_in[8];
    const float* conv_b = (const float*)d_in[9];
    const float* lng    = (const float*)d_in[10];
    const float* lnb    = (const float*)d_in[11];
    float* out = (float*)d_out;

    char* ws = (char*)d_ws;
    int*       idx   = (int*)      (ws + OFF_IDX);
    float*     stats = (float*)    (ws + OFF_STATS);
    float*     img32 = (float*)    (ws + OFF_IMG32);
    _Float16*  img16 = (_Float16*) (ws + OFF_IMG16);
    _Float16*  wt    = (_Float16*) (ws + OFF_WT);
    float*     st    = (float*)    (ws + OFF_ST);

    const int n_img32 = B_*HW_*CIN_;
    k_zero     <<<(n_img32 + 255)/256, 256, 0, stream>>>(img32, n_img32);
    k_zero     <<<1, 64, 0, stream>>>(stats, 8);
    k_ballquery<<<B_*N_/256, 256, 0, stream>>>(xytp, idx);
    k_stats    <<<B_*N_/256, 256, 0, stream>>>(xytp, idx, w1, b1, stats);
    k_scatter  <<<(B_*N_*CIN_ + 255)/256, 256, 0, stream>>>(xytp, feats, img32);
    k_convert  <<<(B_*HW_*CINP_ + 255)/256, 256, 0, stream>>>(img32, img16);
    k_wprep    <<<(NTAP_*COUT_*CINP_ + 255)/256, 256, 0, stream>>>(conv_w, wt);
    k_conv     <<<B_*N_/EVB_, 192, 0, stream>>>(xytp, img16, wt, conv_b, st);
    k_attn     <<<B_*N_/8, 256, 0, stream>>>(xytp, idx, stats, w1, b1, bng, bnb,
                                             w2, b2, st, lng, lnb, out);
}
